// OLMoELayer_5987184410859
// MI455X (gfx1250) — matrix operand order repaired and checked by reading, not compiled, not run on hardware
//
#include <hip/hip_runtime.h>
#include <hip/hip_bf16.h>
#include <stddef.h>

#define Tdim 512
#define Hdim 2048
#define Idim 1024
#define Edim 64
#define KTOP 8
#define MBLK 32            // tokens per workgroup (two 16-row WMMA tiles)

typedef __bf16 bf16_t;
typedef bf16_t v16bf __attribute__((ext_vector_type(16)));
typedef float  v8f   __attribute__((ext_vector_type(8)));
typedef unsigned short us8 __attribute__((ext_vector_type(8)));

union Frag {
    v16bf v;
    unsigned int u[8];
    unsigned short us[16];
    us8 h[2];
};

// Pack two fp32 (bit patterns) into two truncated bf16 in one dword:
// result = (hi16(u1) << 16) | hi16(u0)  -- single v_perm_b32.
__device__ __forceinline__ unsigned pack_bf16(unsigned u0, unsigned u1) {
    return __builtin_amdgcn_perm(u1, u0, 0x07060302u);
}

// fp32 -> bf16 RNE (only for the few activation stores)
__device__ __forceinline__ unsigned short f2bf(float f) {
    union { float f; unsigned u; } c; c.f = f;
    unsigned u = c.u;
    u += 0x7FFFu + ((u >> 16) & 1u);
    return (unsigned short)(u >> 16);
}

// B fragment: 16 consecutive fp32 from one weight row -> 16 bf16.
// 4x global_load_b128 + 8x v_perm_b32.
__device__ __forceinline__ v16bf loadB(const float* __restrict__ p) {
    const uint4* q = (const uint4*)p;
    uint4 a = q[0], b = q[1], c = q[2], d = q[3];
    Frag f;
    f.u[0] = pack_bf16(a.x, a.y); f.u[1] = pack_bf16(a.z, a.w);
    f.u[2] = pack_bf16(b.x, b.y); f.u[3] = pack_bf16(b.z, b.w);
    f.u[4] = pack_bf16(c.x, c.y); f.u[5] = pack_bf16(c.z, c.w);
    f.u[6] = pack_bf16(d.x, d.y); f.u[7] = pack_bf16(d.z, d.w);
    return f.v;
}

// A fragment from a bf16 LDS row: slots 0..7 = cols cbase..cbase+7,
// slots 8..15 = cols cbase+8..cbase+15 (contiguous k order, matches loadB).
__device__ __forceinline__ v16bf loadA_lds(const unsigned short* row, int cbase) {
    Frag a;
    a.h[0] = *(const us8*)(row + cbase);
    a.h[1] = *(const us8*)(row + cbase + 8);
    return a.v;
}

__device__ __forceinline__ v8f wmma_bf16(v16bf a, v16bf b, v8f c) {
    return __builtin_amdgcn_wmma_f32_16x16x32_bf16(
        /*neg_a=*/false, a, /*neg_b=*/false, b,
        /*c_mod=*/(short)0, c, /*reuse_a=*/false, /*reuse_b=*/false);
}

// ---------------------------------------------------------------------------
// Kernel 0: zero output + per-expert counters
// ---------------------------------------------------------------------------
__global__ void zero_kernel(float* __restrict__ out, int nOut,
                            int* __restrict__ counts, int nC) {
    int i = blockIdx.x * blockDim.x + threadIdx.x;
    if (i < nOut) out[i] = 0.0f;
    if (i < nC)   counts[i] = 0;
}

// ---------------------------------------------------------------------------
// Kernel 1: gating — logits, top-8, softmax, build per-expert routing lists
// ---------------------------------------------------------------------------
__global__ void __launch_bounds__(256) gate_kernel(
    const float* __restrict__ x, const float* __restrict__ Wgate,
    int* __restrict__ counts, int* __restrict__ etoks,
    float* __restrict__ eprobs) {
    __shared__ float part[4][Edim];
    __shared__ float logits[Edim];

    const int t   = blockIdx.x;
    const int tid = threadIdx.x;
    const int g   = tid >> 6;   // 0..3 (H partition)
    const int e   = tid & 63;   // expert

    const float* xr = x + (size_t)t * Hdim;
    const float* wr = Wgate + (size_t)e * Hdim;
    float s = 0.0f;
    const int h0 = g * (Hdim / 4);
    for (int h = h0; h < h0 + Hdim / 4; h += 4) {
        float4 xv = *(const float4*)(xr + h);
        float4 wv = *(const float4*)(wr + h);
        s += xv.x * wv.x + xv.y * wv.y + xv.z * wv.z + xv.w * wv.w;
    }
    part[g][e] = s;
    __syncthreads();
    if (tid < Edim)
        logits[tid] = part[0][tid] + part[1][tid] + part[2][tid] + part[3][tid];
    __syncthreads();

    if (tid == 0) {
        float l[Edim];
#pragma unroll
        for (int i = 0; i < Edim; ++i) l[i] = logits[i];
        int   ids[KTOP];
        float vals[KTOP];
        for (int k = 0; k < KTOP; ++k) {
            int bi = 0; float bv = l[0];
            for (int i = 1; i < Edim; ++i)
                if (l[i] > bv) { bv = l[i]; bi = i; }
            ids[k] = bi; vals[k] = bv; l[bi] = -3.4e38f;
        }
        float mx = vals[0];                // vals sorted descending
        float p[KTOP]; float sum = 0.0f;
        for (int k = 0; k < KTOP; ++k) { p[k] = __expf(vals[k] - mx); sum += p[k]; }
        float inv = __builtin_amdgcn_rcpf(sum);
        for (int k = 0; k < KTOP; ++k) {
            int ex   = ids[k];
            int slot = atomicAdd(&counts[ex], 1);
            etoks [ex * Tdim + slot] = t;
            eprobs[ex * Tdim + slot] = p[k] * inv;
        }
    }
}

// ---------------------------------------------------------------------------
// Kernel 2: per-(expert, 32-token tile) SwiGLU + down-proj, bf16 WMMA.
// Each wave computes TWO 16-row M-tiles against one shared B fragment:
// every weight byte loaded from global feeds 2 WMMAs.
// Dynamic LDS: xs[32*Hdim] bf16 (128 KB) | acts[32*Idim] bf16 (64 KB).
// ---------------------------------------------------------------------------
__global__ void __launch_bounds__(256) expert_kernel(
    const float* __restrict__ x,  const float* __restrict__ Wg,
    const float* __restrict__ Wu, const float* __restrict__ Wd,
    const int* __restrict__ counts, const int* __restrict__ etoks,
    const float* __restrict__ eprobs, float* __restrict__ out) {
    extern __shared__ unsigned short smem[];
    unsigned short* xs   = smem;                 // MBLK x Hdim bf16
    unsigned short* acts = smem + MBLK * Hdim;   // MBLK x Idim bf16
    __shared__ float probs[MBLK];
    __shared__ int   toks[MBLK];

    const int tile = blockIdx.x;     // tile-fastest: twins of an expert adjacent
    const int e    = blockIdx.y;
    const int cnt  = counts[e];
    if (tile * MBLK >= cnt) return;

    const int tid  = threadIdx.x;
    const int wave = tid >> 5;     // 0..7
    const int lane = tid & 31;
    const int m16  = lane & 15;    // A row / B,D column within 16-tile
    const int hi   = lane >> 4;    // lane group (K-half select)

    if (tid < MBLK) {
        int idx = tile * MBLK + tid;
        int tk = -1; float p = 0.0f;
        if (idx < cnt) { tk = etoks[e * Tdim + idx]; p = eprobs[e * Tdim + idx]; }
        toks[tid] = tk; probs[tid] = p;
    }
    __syncthreads();

    // Stage x tile (MBLK rows x Hdim) into LDS as bf16. Padded rows (tok<0)
    // duplicate row 0's token (always valid: tile*MBLK < cnt); their prob==0
    // zeroes the activation, and the output atomic is guarded by tok>=0.
    for (int v = tid; v < (MBLK * Hdim) / 8; v += 256) {
        const int row = v >> 8;               // Hdim/8 = 256 chunks per row
        const int col = (v & 255) * 8;
        int tk = toks[row]; if (tk < 0) tk = toks[0];
        const uint4* q = (const uint4*)(x + (size_t)tk * Hdim + col);
        uint4 a = q[0], b = q[1];
        uint4 r;
        r.x = pack_bf16(a.x, a.y); r.y = pack_bf16(a.z, a.w);
        r.z = pack_bf16(b.x, b.y); r.w = pack_bf16(b.z, b.w);
        *(uint4*)(xs + (size_t)row * Hdim + col) = r;
    }
    __syncthreads();

    // Per-lane register copies of token/prob metadata for both M-tiles.
    float pj0[8], pj1[8]; int tk0[8], tk1[8];
#pragma unroll
    for (int j = 0; j < 8; ++j) {
        const int m = j + hi * 8;
        pj0[j] = probs[m];      tk0[j] = toks[m];
        pj1[j] = probs[16 + m]; tk1[j] = toks[16 + m];
    }

    const float* Wge = Wg + (size_t)e * Idim * Hdim;
    const float* Wue = Wu + (size_t)e * Idim * Hdim;
    const float* Wde = Wd + (size_t)e * Hdim * Idim;
    const unsigned short* xrow0 = xs + (size_t)m16 * Hdim;
    const unsigned short* xrow1 = xs + (size_t)(16 + m16) * Hdim;

    // ---- Phase 1: hg/hu = x @ Wg^T, x @ Wu^T ; act = silu(hg)*hu*prob ----
    for (int ic = 0; ic < Idim; ic += 128) {
        const int i0 = ic + wave * 16;
        v8f accg0 = {0.f,0.f,0.f,0.f,0.f,0.f,0.f,0.f};
        v8f accu0 = {0.f,0.f,0.f,0.f,0.f,0.f,0.f,0.f};
        v8f accg1 = {0.f,0.f,0.f,0.f,0.f,0.f,0.f,0.f};
        v8f accu1 = {0.f,0.f,0.f,0.f,0.f,0.f,0.f,0.f};
        const float* grow = Wge + (size_t)(i0 + m16) * Hdim + hi * 16;
        const float* urow = Wue + (size_t)(i0 + m16) * Hdim + hi * 16;
        for (int h = 0; h < Hdim; h += 32) {
            v16bf bg = loadB(grow + h);            // one weight fragment ...
            v16bf bu = loadB(urow + h);
            v16bf a0 = loadA_lds(xrow0, h + hi * 16);
            v16bf a1 = loadA_lds(xrow1, h + hi * 16);
            accg0 = wmma_bf16(a0, bg, accg0);      // ... feeds two WMMAs
            accg1 = wmma_bf16(a1, bg, accg1);
            accu0 = wmma_bf16(a0, bu, accu0);
            accu1 = wmma_bf16(a1, bu, accu1);
        }
#pragma unroll
        for (int j = 0; j < 8; ++j) {
            const int m = j + hi * 8;
            float g0 = accg0[j], u0 = accu0[j];
            float g1 = accg1[j], u1 = accu1[j];
            float s0 = __builtin_amdgcn_rcpf(1.0f + __expf(-g0));
            float s1 = __builtin_amdgcn_rcpf(1.0f + __expf(-g1));
            acts[m        * Idim + i0 + m16] = f2bf(g0 * s0 * u0 * pj0[j]);
            acts[(16 + m) * Idim + i0 + m16] = f2bf(g1 * s1 * u1 * pj1[j]);
        }
    }
    __syncthreads();

    // ---- Phase 2: out[t, :] += act @ Wd[e]^T (atomic across experts) ----
    const unsigned short* arow0 = acts + (size_t)m16 * Idim;
    const unsigned short* arow1 = acts + (size_t)(16 + m16) * Idim;
    for (int hc = 0; hc < Hdim; hc += 128) {
        const int h0 = hc + wave * 16;
        v8f acc0 = {0.f,0.f,0.f,0.f,0.f,0.f,0.f,0.f};
        v8f acc1 = {0.f,0.f,0.f,0.f,0.f,0.f,0.f,0.f};
        const float* drow = Wde + (size_t)(h0 + m16) * Idim + hi * 16;
        for (int i = 0; i < Idim; i += 32) {
            v16bf b  = loadB(drow + i);            // shared Wd fragment
            v16bf a0 = loadA_lds(arow0, i + hi * 16);
            v16bf a1 = loadA_lds(arow1, i + hi * 16);
            acc0 = wmma_bf16(a0, b, acc0);
            acc1 = wmma_bf16(a1, b, acc1);
        }
#pragma unroll
        for (int j = 0; j < 8; ++j) {
            if (tk0[j] >= 0)
                atomicAdd(out + (size_t)tk0[j] * Hdim + h0 + m16, acc0[j]);
            if (tk1[j] >= 0)
                atomicAdd(out + (size_t)tk1[j] * Hdim + h0 + m16, acc1[j]);
        }
    }
}

// ---------------------------------------------------------------------------
extern "C" void kernel_launch(void* const* d_in, const int* in_sizes, int n_in,
                              void* d_out, int out_size, void* d_ws, size_t ws_size,
                              hipStream_t stream) {
    const float* x     = (const float*)d_in[0];
    const float* Wg    = (const float*)d_in[1];
    const float* Wu    = (const float*)d_in[2];
    const float* Wd    = (const float*)d_in[3];
    const float* Wgate = (const float*)d_in[4];
    // d_in[5] = k (== 8), compile-time constant KTOP here.
    float* out = (float*)d_out;

    // workspace layout: counts[E] | etoks[E*T] | eprobs[E*T]
    int*   counts = (int*)d_ws;
    int*   etoks  = counts + Edim;
    float* eprobs = (float*)(etoks + (size_t)Edim * Tdim);

    const int nOut = Tdim * Hdim;
    zero_kernel<<<(nOut + 255) / 256, 256, 0, stream>>>(out, nOut, counts, Edim);
    gate_kernel<<<Tdim, 256, 0, stream>>>(x, Wgate, counts, etoks, eprobs);

    const size_t shmem = (size_t)(MBLK * Hdim + MBLK * Idim) * sizeof(unsigned short);
    dim3 grid(Tdim / MBLK, Edim);    // tile-fastest so an expert's twin
    expert_kernel<<<grid, 256, shmem, stream>>>(x, Wg, Wu, Wd,   // workgroups co-travel in L2
                                                counts, etoks, eprobs, out);
}
